// GCNsNet_7112465842805
// MI455X (gfx1250) — matrix operand order repaired and checked by reading, not compiled, not run on hardware
//
#include <hip/hip_runtime.h>
#include <cstdint>
#include <cstddef>

#define N_NODES 30000
#define N_EDGES 480000

// ---------------- WMMA types (CDNA5 / gfx1250, wave32) ----------------
typedef __bf16 bf16;
typedef bf16  v16bf __attribute__((ext_vector_type(16)));
typedef float v8f   __attribute__((ext_vector_type(8)));

union FragAB {
    v16bf v;
    uint4 q[2];
};

// ---------------- TDM (Tensor Data Mover) support ----------------
#if defined(__AMDGCN__) && __has_builtin(__builtin_amdgcn_tensor_load_to_lds)
#define HAVE_TDM 1
#else
#define HAVE_TDM 0
#endif

#if HAVE_TDM
typedef unsigned int u32x4 __attribute__((ext_vector_type(4)));
typedef int i32x4 __attribute__((ext_vector_type(4)));
typedef int i32x8 __attribute__((ext_vector_type(8)));

// 2D tile load, 2-byte elements, LDS row padding 64B data + 16B pad (stride 80B).
// OOB (beyond tensor_d0/tensor_d1) reads return zero per ISA sec 8.2.
__device__ __forceinline__ void tdm_load_tile_2d(
    unsigned lds_addr, const void* gaddr,
    unsigned tensor_d0, unsigned tensor_d1,
    unsigned tile_d0, unsigned tile_d1, unsigned stride0) {
    unsigned long long ga = (unsigned long long)gaddr;
    u32x4 g0;
    g0[0] = 1u;                                    // count=1 (valid), user mode
    g0[1] = lds_addr;                              // D#.lds_addr (bytes)
    g0[2] = (unsigned)ga;                          // global_addr[31:0]
    g0[3] = (unsigned)((ga >> 32) & 0x01ffffffu)   // global_addr[56:32]
            | (2u << 30);                          // D#.type = 2 ("image")
    i32x8 g1;
    g1[0] = (int)((1u << 16)    // data_size = 1 -> 2 bytes
            | (1u << 20)        // pad_enable
            | (3u << 22)        // pad_interval code 3 -> 16 DWORDs (64B of data)
            | (3u << 25));      // pad_amount code 3 -> 4 DWORDs (16B pad)
    g1[1] = (int)((tensor_d0 & 0xffffu) << 16);                       // tensor_dim0 lo16
    g1[2] = (int)((tensor_d0 >> 16) | ((tensor_d1 & 0xffffu) << 16)); // d0 hi16 | d1 lo16
    g1[3] = (int)((tensor_d1 >> 16) | (tile_d0 << 16));               // d1 hi16 | tile_dim0
    g1[4] = (int)(tile_d1 & 0xffffu);                                 // tile_dim1; tile_dim2=0
    g1[5] = (int)stride0;                                             // tensor_dim0_stride lo32
    g1[6] = 0;
    g1[7] = 0;
    i32x4 z4 = {0, 0, 0, 0};
#if __clang_major__ >= 23
    i32x8 z8 = {0, 0, 0, 0, 0, 0, 0, 0};
    __builtin_amdgcn_tensor_load_to_lds(g0, g1, z4, z4, z8, 0);
#else
    __builtin_amdgcn_tensor_load_to_lds(g0, g1, z4, z4, 0);
#endif
}
#endif  // HAVE_TDM

// ---------------- small utility kernels ----------------
__global__ void fill_zero4(float4* __restrict__ p, int n4) {
    int i = blockIdx.x * blockDim.x + threadIdx.x;
    if (i < n4) p[i] = float4{0.f, 0.f, 0.f, 0.f};
}

__global__ void neg_copy4(float4* __restrict__ dst, const float4* __restrict__ src, int n4) {
    int i = blockIdx.x * blockDim.x + threadIdx.x;
    if (i < n4) {
        float4 v = src[i];
        dst[i] = float4{-v.x, -v.y, -v.z, -v.w};
    }
}

__global__ void deg_scatter(const int* __restrict__ row, const float* __restrict__ ew,
                            float* __restrict__ deg, int ne) {
    int e = blockIdx.x * blockDim.x + threadIdx.x;
    if (e < ne) unsafeAtomicAdd(&deg[row[e]], ew[e]);
}

__global__ void woff_kernel(const int* __restrict__ row, const float* __restrict__ ew,
                            const float* __restrict__ deg, float* __restrict__ woff, int ne) {
    int e = blockIdx.x * blockDim.x + threadIdx.x;
    if (e < ne) {
        float d = deg[row[e]];
        float dinv = (d > 0.f) ? (1.f / d) : 0.f;
        woff[e] = -dinv * ew[e];
    }
}

// out[col[e], :] += scale * woff[e] * v[row[e], :]   (float4 per thread, L2-resident atomics)
__global__ void spmv_scatter(const int* __restrict__ row, const int* __restrict__ col,
                             const float* __restrict__ woff, const float* __restrict__ v,
                             float* __restrict__ out, int F, int f4shift, float scale, int total) {
    int tid = blockIdx.x * blockDim.x + threadIdx.x;
    if (tid >= total) return;
    int e = tid >> f4shift;
    int c = tid & ((1 << f4shift) - 1);
    int r  = row[e];
    int cc = col[e];
    float w = woff[e] * scale;
    const float4 xv = *(const float4*)(v + (size_t)r * F + 4 * c);
    float* o = out + (size_t)cc * F + 4 * c;
    unsafeAtomicAdd(o + 0, w * xv.x);
    unsafeAtomicAdd(o + 1, w * xv.y);
    unsafeAtomicAdd(o + 2, w * xv.z);
    unsafeAtomicAdd(o + 3, w * xv.w);
}

// Convert three f32 matrices (same shape) to bf16; blockIdx.y selects the matrix.
__global__ void cvt3_bf16(const float* __restrict__ s0, const float* __restrict__ s1,
                          const float* __restrict__ s2, bf16* __restrict__ d0,
                          bf16* __restrict__ d1, bf16* __restrict__ d2, int n) {
    const float* s = (blockIdx.y == 0) ? s0 : ((blockIdx.y == 1) ? s1 : s2);
    bf16* d = (blockIdx.y == 0) ? d0 : ((blockIdx.y == 1) ? d1 : d2);
    int i = blockIdx.x * blockDim.x + threadIdx.x;   // 8 elements per thread
    if (8 * i < n) {
        const float4* sp = (const float4*)(s + 8 * (size_t)i);
        float4 u0 = sp[0], u1 = sp[1];
        bf16 tmp[8];
        tmp[0] = (bf16)u0.x; tmp[1] = (bf16)u0.y; tmp[2] = (bf16)u0.z; tmp[3] = (bf16)u0.w;
        tmp[4] = (bf16)u1.x; tmp[5] = (bf16)u1.y; tmp[6] = (bf16)u1.z; tmp[7] = (bf16)u1.w;
        *(uint4*)(d + 8 * (size_t)i) = *(const uint4*)tmp;
    }
}

// Wt[kc][n][k] = (bf16) W[kc][k][n]   (weights -> column-major bf16, coalesced writes)
__global__ void wcvt_t(const float* __restrict__ W, bf16* __restrict__ Wt, int Fi, int Fo) {
    int i = blockIdx.x * blockDim.x + threadIdx.x;
    int tot = 3 * Fi * Fo;
    if (i < tot) {
        int kc  = i / (Fi * Fo);
        int rem = i - kc * Fi * Fo;
        int n   = rem / Fi;
        int k   = rem - n * Fi;
        Wt[i] = (bf16)W[(size_t)kc * Fi * Fo + (size_t)k * Fo + n];
    }
}

// ---------------- fused Chebyshev GEMM + bias + softplus (bf16 WMMA) ----------------
// out[N x Fo] = softplus( A0@W0 + A1@W1 + A2@W2 + bias )
// BM=64, BN=128, BK=32 ; 256 threads = 8 waves (2x4), wave -> 32x32 via 2x2 WMMA tiles.
// TDM path: double-buffered LDS, wave 0 issues tile-pair s+1 then waits TENSORcnt<=2
// (pairs retire in order) so DMA for the next tile overlaps WMMA on the current one.
#define BM 64
#define BN 128
#define BK 32
#define LDA 40    // As row stride (bf16): 64B data + 16B pad = TDM pad config
#define LDB 40    // Bt col stride (bf16)

__global__ __launch_bounds__(256) void cheb_gemm_wmma(
    const bf16* __restrict__ A0, const bf16* __restrict__ A1, const bf16* __restrict__ A2,
    const bf16* __restrict__ Wt,   // (3, Fo, Fi) bf16 column-major
    const float* __restrict__ bias,
    float* __restrict__ out, int Fi, int Fo, int n_rows) {

    __shared__ bf16 As[2][BM * LDA];   // 64 rows x 32 k (+pad), double buffered
    __shared__ bf16 Bt[2][BN * LDB];   // 128 cols x 32 k (transposed, +pad), double buffered

    const int tid  = threadIdx.x;
    const int wid  = tid >> 5;
    const int lane = tid & 31;
    const int waveM = wid >> 2;     // 0..1
    const int waveN = wid & 3;      // 0..3
    const int half  = lane >> 4;    // 0/1
    const int l15   = lane & 15;

    const int rowBlock = blockIdx.x * BM;
    const int colBlock = blockIdx.y * BN;

    const int ksteps = (Fi + BK - 1) / BK;
    const int nsteps = 3 * ksteps;

    v8f acc[2][2] = {};
    const bf16* Amats[3] = {A0, A1, A2};

#if HAVE_TDM
    auto issue_pair = [&](int s, int buf) {
        int kc = s / ksteps;
        int k0 = (s - kc * ksteps) * BK;
        tdm_load_tile_2d((unsigned)(size_t)(void*)As[buf],
                         Amats[kc] + (size_t)rowBlock * Fi + k0,
                         (unsigned)(Fi - k0), (unsigned)(n_rows - rowBlock),
                         BK, BM, (unsigned)Fi);
        tdm_load_tile_2d((unsigned)(size_t)(void*)Bt[buf],
                         Wt + ((size_t)kc * Fo + colBlock) * (size_t)Fi + k0,
                         (unsigned)(Fi - k0), (unsigned)(Fo - colBlock),
                         BK, BN, (unsigned)Fi);
    };
    if (wid == 0) issue_pair(0, 0);
#else
    // manual staging fallback (also what the host pass sees)
    const int arow   = tid >> 2;            // 0..63
    const int achunk = (tid & 3) << 3;      // k offset: 0,8,16,24
    const int bcol   = tid >> 1;            // 0..127
    const int bchunk = (tid & 1) << 4;      // k offset: 0,16
    const int growc  = min(rowBlock + arow, n_rows - 1);
    const int gcolc  = min(colBlock + bcol, Fo - 1);
#endif

    int p = 0;
    for (int s = 0; s < nsteps; ++s) {
#if HAVE_TDM
        if (wid == 0) {
            if (s + 1 < nsteps) {
                issue_pair(s + 1, p ^ 1);   // prefetch next tile pair into other buffer
                __builtin_amdgcn_s_wait_tensorcnt(2);  // pair s complete (in-order retire)
            } else {
                __builtin_amdgcn_s_wait_tensorcnt(0);
            }
        }
#else
        {
            int kc = s / ksteps;
            int k0 = (s - kc * ksteps) * BK;
            const bf16* A = Amats[kc];
            {   // A tile 64x32: one b128 per thread; zero k-overhang
                bool kok = (k0 + achunk + 8) <= Fi;
                int  ka  = kok ? (k0 + achunk) : 0;
                uint4 val = *(const uint4*)(A + (size_t)growc * Fi + ka);
                if (!kok) val = uint4{0u, 0u, 0u, 0u};
                *(uint4*)&As[p][arow * LDA + achunk] = val;
            }
            {   // B tile 128x32 transposed: two b128 per thread; clamp only
                const bf16* wcol = Wt + ((size_t)kc * Fo + gcolc) * (size_t)Fi;
                bool kok = (k0 + bchunk + 16) <= Fi;
                int  kb  = kok ? (k0 + bchunk) : 0;
                const uint4* pp = (const uint4*)(wcol + kb);
                uint4 b0 = pp[0];
                uint4 b1 = pp[1];
                *(uint4*)&Bt[p][bcol * LDB + bchunk]     = b0;
                *(uint4*)&Bt[p][bcol * LDB + bchunk + 8] = b1;
            }
        }
#endif
        __syncthreads();

        // ---- fragments: all b128 LDS loads (ISA 7.12.2 16-bit layouts) ----
        FragAB a[2], b[2];
#pragma unroll
        for (int tm = 0; tm < 2; ++tm) {
            int r = waveM * 32 + tm * 16 + l15;
            a[tm].q[0] = *(const uint4*)&As[p][r * LDA + 8 * half];        // k = 8h .. 8h+7
            a[tm].q[1] = *(const uint4*)&As[p][r * LDA + 16 + 8 * half];   // k = 16+8h .. +7
        }
#pragma unroll
        for (int tn = 0; tn < 2; ++tn) {
            int c = waveN * 32 + tn * 16 + l15;
            b[tn].q[0] = *(const uint4*)&Bt[p][c * LDB + 8 * half];        // k = 8h .. 8h+7
            b[tn].q[1] = *(const uint4*)&Bt[p][c * LDB + 16 + 8 * half];   // k = 16+8h .. +7
        }
#pragma unroll
        for (int tm = 0; tm < 2; ++tm)
#pragma unroll
            for (int tn = 0; tn < 2; ++tn)
                acc[tm][tn] = __builtin_amdgcn_wmma_f32_16x16x32_bf16(
                    false, a[tm].v, false, b[tn].v, (short)0, acc[tm][tn], false, false);
        __syncthreads();
        p ^= 1;
    }

    // ---- epilogue: bias + softplus; C layout: VGPR r -> M = r + 8*half, N = lane%16 ----
#pragma unroll
    for (int tm = 0; tm < 2; ++tm) {
#pragma unroll
        for (int tn = 0; tn < 2; ++tn) {
            int gcol = colBlock + waveN * 32 + tn * 16 + l15;
            if (gcol >= Fo) continue;
            float bv = bias[gcol];
#pragma unroll
            for (int r = 0; r < 8; ++r) {
                int grow = rowBlock + waveM * 32 + tm * 16 + r + 8 * half;
                if (grow < n_rows) {
                    float vo = acc[tm][tn][r] + bv;
                    float sp = (vo > 15.f) ? vo : __logf(1.f + __expf(vo));
                    out[(size_t)grow * Fo + gcol] = sp;
                }
            }
        }
    }
}

// ---------------- final FC: (N x 512) @ (512 x 3) + b ----------------
__global__ void fc_kernel(const float* __restrict__ h, const float* __restrict__ w,
                          const float* __restrict__ b, float* __restrict__ out, int n) {
    int tid = blockIdx.x * blockDim.x + threadIdx.x;
    if (tid >= n * 3) return;
    int node = tid / 3;
    int j = tid - node * 3;
    float acc = b[j];
    const float4* hv = (const float4*)(h + (size_t)node * 512);
#pragma unroll 4
    for (int i = 0; i < 128; ++i) {
        float4 xv = hv[i];
        acc += xv.x * w[(4 * i + 0) * 3 + j];
        acc += xv.y * w[(4 * i + 1) * 3 + j];
        acc += xv.z * w[(4 * i + 2) * 3 + j];
        acc += xv.w * w[(4 * i + 3) * 3 + j];
    }
    out[tid] = acc;
}

// ---------------- launcher ----------------
extern "C" void kernel_launch(void* const* d_in, const int* in_sizes, int n_in,
                              void* d_out, int out_size, void* d_ws, size_t ws_size,
                              hipStream_t stream) {
    (void)in_sizes; (void)n_in; (void)out_size; (void)ws_size;

    const float* x    = (const float*)d_in[0];
    const int*   eidx = (const int*)d_in[1];
    const float* ew   = (const float*)d_in[2];
    // d_in[3] = batch (identity pooling) -> unused
    const float* W[6];
    const float* B[6];
    for (int i = 0; i < 6; ++i) {
        W[i] = (const float*)d_in[4 + 2 * i];
        B[i] = (const float*)d_in[5 + 2 * i];
    }
    const float* fcw = (const float*)d_in[16];
    const float* fcb = (const float*)d_in[17];

    const int* row = eidx;
    const int* col = eidx + N_EDGES;

    // workspace carve-out
    char* ws = (char*)d_ws;
    size_t off = 0;
    auto carve = [&](size_t bytes) -> void* {
        void* p = ws + off;
        off += (bytes + 255) & ~(size_t)255;
        return p;
    };
    float* deg  = (float*)carve((size_t)N_NODES * 4);
    float* woff = (float*)carve((size_t)N_EDGES * 4);
    float* hA   = (float*)carve((size_t)N_NODES * 256 * 4);  // outputs of layers 1,3,5 (Fo<=256)
    float* hB   = (float*)carve((size_t)N_NODES * 512 * 4);  // outputs of layers 2,4,6 (Fo<=512)
    float* tx1  = (float*)carve((size_t)N_NODES * 256 * 4);
    float* tx2  = (float*)carve((size_t)N_NODES * 256 * 4);
    bf16*  a0b  = (bf16*)carve((size_t)N_NODES * 256 * 2);
    bf16*  a1b  = (bf16*)carve((size_t)N_NODES * 256 * 2);
    bf16*  a2b  = (bf16*)carve((size_t)N_NODES * 256 * 2);
    bf16*  wtb  = (bf16*)carve((size_t)3 * 256 * 512 * 2);

    // ---- Laplacian normalization ----
    {
        int n4 = N_NODES / 4;
        fill_zero4<<<(n4 + 255) / 256, 256, 0, stream>>>((float4*)deg, n4);
    }
    deg_scatter<<<(N_EDGES + 255) / 256, 256, 0, stream>>>(row, ew, deg, N_EDGES);
    woff_kernel<<<(N_EDGES + 255) / 256, 256, 0, stream>>>(row, ew, deg, woff, N_EDGES);

    const int fin[6]  = {128, 16, 32, 64, 128, 256};
    const int fout[6] = {16, 32, 64, 128, 256, 512};

    const float* cur = x;
    float* nxt = hA;
    for (int l = 0; l < 6; ++l) {
        int Fi = fin[l], Fo = fout[l];
        int n   = N_NODES * Fi;
        int n4  = n / 4;
        int f4  = Fi / 4;
        int shift = __builtin_ctz(f4);
        int total = N_EDGES * f4;

        // Tx1 = Lhat @ cur
        fill_zero4<<<(n4 + 255) / 256, 256, 0, stream>>>((float4*)tx1, n4);
        spmv_scatter<<<(total + 255) / 256, 256, 0, stream>>>(row, col, woff, cur, tx1,
                                                              Fi, shift, 1.0f, total);
        // Tx2 = 2 * Lhat @ Tx1 - cur
        neg_copy4<<<(n4 + 255) / 256, 256, 0, stream>>>((float4*)tx2, (const float4*)cur, n4);
        spmv_scatter<<<(total + 255) / 256, 256, 0, stream>>>(row, col, woff, tx1, tx2,
                                                              Fi, shift, 2.0f, total);

        // bf16 operand prep (activations + transposed weights)
        dim3 cg((n / 8 + 255) / 256, 3);
        cvt3_bf16<<<cg, 256, 0, stream>>>(cur, tx1, tx2, a0b, a1b, a2b, n);
        int wtot = 3 * Fi * Fo;
        wcvt_t<<<(wtot + 255) / 256, 256, 0, stream>>>(W[l], wtb, Fi, Fo);

        // out = softplus(cur@W0 + Tx1@W1 + Tx2@W2 + b)
        dim3 grid((N_NODES + BM - 1) / BM, (Fo + BN - 1) / BN);
        cheb_gemm_wmma<<<grid, 256, 0, stream>>>(a0b, a1b, a2b, wtb, B[l], nxt, Fi, Fo, N_NODES);

        cur = nxt;
        nxt = (cur == hA) ? hB : hA;
    }

    fc_kernel<<<(N_NODES * 3 + 255) / 256, 256, 0, stream>>>(cur, fcw, fcb, (float*)d_out, N_NODES);
}
